// PyramidResidualMoE_56693568307687
// MI455X (gfx1250) — compile-verified, matrix operand order checked by reading, not run on hardware
//
#include <hip/hip_runtime.h>
#include <hip/hip_bf16.h>

typedef __attribute__((ext_vector_type(16))) __bf16 v16bf;
typedef __attribute__((ext_vector_type(8)))  float  v8f;
typedef unsigned short u16;

#define NTOK   (8 * 2048)   // B*T
#define CDIM   896
#define TM     128          // token-tile (rows)
#define TNW    128          // output-column tile
#define LDST   40           // LDS row stride in u16 elems (32 data + 8 pad)
#define BUFB   (TM * LDST * 2)  // byte stride between double buffers

// ---------- helpers ----------

__device__ __forceinline__ u16 f32_to_bf16(float f) {
  unsigned int u = __float_as_uint(f);
  unsigned int r = (u + 0x7fffu + ((u >> 16) & 1u)) >> 16;
  return (u16)r;
}

__device__ __forceinline__ v16bf ld_frag(const u16* p) {
  // p is 16B aligned; fragment = 8 elems at p, 8 elems at p+16 (ISA 16-bit A/B layout)
  union { v16bf v; uint4 q[2]; } u;
  u.q[0] = *(const uint4*)(p);
  u.q[1] = *(const uint4*)(p + 16);
  return u.v;
}

__device__ __forceinline__ float fast_sigmoid(float z) {
  return __builtin_amdgcn_rcpf(1.0f + __expf(-z));
}

// LDS byte offset (group-segment relative) of a generic pointer into __shared__
__device__ __forceinline__ unsigned lds_off(const void* p) {
  return (unsigned)(size_t)(__attribute__((address_space(3))) const void*)p;
}

// CDNA5 async memory->LDS copy, tracked by ASYNCcnt (no VGPR data path)
__device__ __forceinline__ void async_b128(unsigned lds_addr, const void* gptr) {
  asm volatile("global_load_async_to_lds_b128 %0, %1, off"
               :: "v"(lds_addr), "v"(gptr) : "memory");
}
__device__ __forceinline__ void wait_async0() {
  asm volatile("s_wait_asynccnt 0x0" ::: "memory");
}

// ---------- elementwise kernels ----------

__global__ void cvt_f32_bf16(const float* __restrict__ in, u16* __restrict__ out, int n) {
  for (int i = blockIdx.x * blockDim.x + threadIdx.x; i < n; i += gridDim.x * blockDim.x)
    out[i] = f32_to_bf16(in[i]);
}

// convert [H,C] -> [Hpad,C]: zero tail rows
__global__ void cvt_pad_tail(const float* __restrict__ in, u16* __restrict__ out,
                             int nvalid, int ntotal) {
  for (int i = blockIdx.x * blockDim.x + threadIdx.x; i < ntotal; i += gridDim.x * blockDim.x)
    out[i] = (i < nvalid) ? f32_to_bf16(in[i]) : (u16)0;
}

// convert [R,H] -> [R,Hpad]: zero pad columns
__global__ void cvt_pad_cols(const float* __restrict__ in, u16* __restrict__ out,
                             int rows, int H, int Hpad) {
  int ntot = rows * Hpad;
  for (int i = blockIdx.x * blockDim.x + threadIdx.x; i < ntot; i += gridDim.x * blockDim.x) {
    int r = i / Hpad, c = i - r * Hpad;
    out[i] = (c < H) ? f32_to_bf16(in[r * H + c]) : (u16)0;
  }
}

__global__ void copy_f32(const float* __restrict__ in, float* __restrict__ out, int n) {
  for (int i = blockIdx.x * blockDim.x + threadIdx.x; i < n; i += gridDim.x * blockDim.x)
    out[i] = in[i];
}

__global__ void zero_i32(int* __restrict__ p, int n) {
  for (int i = blockIdx.x * blockDim.x + threadIdx.x; i < n; i += gridDim.x * blockDim.x)
    p[i] = 0;
}

// ---------- router: one wave per token; pushes compacted per-expert lists ----------

__global__ __launch_bounds__(256) void router_kernel(
    const float* __restrict__ x, const float* __restrict__ rw,
    const float* __restrict__ bias, int* __restrict__ counts,
    int* __restrict__ eidx, float* __restrict__ escale, int nt) {
  int gid  = blockIdx.x * blockDim.x + threadIdx.x;
  int t    = gid >> 5;
  int lane = gid & 31;
  if (t >= nt) return;
  const float* xr = x + (size_t)t * CDIM;
  float d0 = 0.f, d1 = 0.f, d2 = 0.f;
  for (int c = lane; c < CDIM; c += 32) {
    float xv = xr[c];
    d0 += xv * rw[c];
    d1 += xv * rw[CDIM + c];
    d2 += xv * rw[2 * CDIM + c];
  }
  #pragma unroll
  for (int off = 16; off; off >>= 1) {
    d0 += __shfl_xor(d0, off, 32);
    d1 += __shfl_xor(d1, off, 32);
    d2 += __shfl_xor(d2, off, 32);
  }
  if (lane == 0) {
    float p0 = fast_sigmoid(d0 + bias[0]);
    float p1 = fast_sigmoid(d1 + bias[1]);
    float p2 = fast_sigmoid(d2 + bias[2]);
    int top = 0; float best = p0;
    if (p1 > best) { best = p1; top = 1; }
    if (p2 > best) { best = p2; top = 2; }
    // after top-1 cap + no-active fallback: active == onehot(top), weight = p/max(p,1e-6)
    float w = best / fmaxf(best, 1e-6f);
    int slot = atomicAdd(&counts[top], 1);
    eidx[top * NTOK + slot]   = t;
    escale[top * NTOK + slot] = w;
  }
}

// ---------- GEMM1: hidden[slot] = silu(x[idx] Wg^T) * (x[idx] Wu^T), bf16 ----------
// A rows gathered via idxp (nullptr => identity / full NTOK).

__global__ __launch_bounds__(256) void gemm_swiglu(
    const u16* __restrict__ xb, const u16* __restrict__ gw,
    const u16* __restrict__ uw, u16* __restrict__ hid,
    const int* __restrict__ idxp, const int* __restrict__ cntp, int Hpad) {
  __shared__ alignas(16) u16 lA[2 * TM * LDST];
  __shared__ alignas(16) u16 lG[2 * TNW * LDST];
  __shared__ alignas(16) u16 lU[2 * TNW * LDST];
  __shared__ int lIdx[TM];

  const int tid    = threadIdx.x;
  const int lane   = tid & 31;
  const int wave   = tid >> 5;
  const int wm     = wave & 3;   // 32-row sub-tile
  const int wn     = wave >> 2;  // 64-col sub-tile
  const int m0     = blockIdx.y * TM;
  const int n0     = blockIdx.x * TNW;
  const int lane15 = lane & 15;
  const int kl     = (lane < 16) ? 0 : 8;

  const int cnt = cntp ? *cntp : NTOK;
  if (m0 >= cnt) return;

  if (tid < TM) lIdx[tid] = idxp ? idxp[m0 + tid] : (m0 + tid);
  __syncthreads();

  // hoisted per-thread fill pointers (rows fixed; only k0 varies)
  const u16* pA[2]; const u16* pG[2]; const u16* pU[2];
  unsigned dA[2], dG[2], dU[2];
  #pragma unroll
  for (int i = 0; i < 2; ++i) {
    int chunk = tid + 256 * i;
    int r = chunk >> 2, j = chunk & 3;
    pA[i] = xb + (size_t)lIdx[r] * CDIM + 8 * j;
    pG[i] = gw + (size_t)(n0 + r) * CDIM + 8 * j;
    pU[i] = uw + (size_t)(n0 + r) * CDIM + 8 * j;
    dA[i] = lds_off(lA + r * LDST + 8 * j);
    dG[i] = lds_off(lG + r * LDST + 8 * j);
    dU[i] = lds_off(lU + r * LDST + 8 * j);
  }

  auto fill = [&](int buf, int k0) {
    unsigned bo = buf * BUFB;
    #pragma unroll
    for (int i = 0; i < 2; ++i) {
      async_b128(dA[i] + bo, pA[i] + k0);
      async_b128(dG[i] + bo, pG[i] + k0);
      async_b128(dU[i] + bo, pU[i] + k0);
    }
  };

  v8f cg[2][4], cu[2][4];
  #pragma unroll
  for (int i = 0; i < 2; ++i)
    #pragma unroll
    for (int j = 0; j < 4; ++j) { cg[i][j] = (v8f)0.0f; cu[i][j] = (v8f)0.0f; }

  const int KIT = CDIM / 32;
  fill(0, 0);
  for (int it = 0; it < KIT; ++it) {
    wait_async0();       // my fill of buffer it&1 complete
    __syncthreads();     // everyone's fill complete; prior reads of other buf done
    if (it + 1 < KIT) fill((it + 1) & 1, 32 * (it + 1));

    const u16* bA = lA + (it & 1) * TM * LDST;
    const u16* bG = lG + (it & 1) * TNW * LDST;
    const u16* bU = lU + (it & 1) * TNW * LDST;
    v16bf a[2], bg[4], bu[4];
    #pragma unroll
    for (int i = 0; i < 2; ++i)
      a[i] = ld_frag(bA + (32 * wm + 16 * i + lane15) * LDST + kl);
    #pragma unroll
    for (int j = 0; j < 4; ++j) {
      bg[j] = ld_frag(bG + (64 * wn + 16 * j + lane15) * LDST + kl);
      bu[j] = ld_frag(bU + (64 * wn + 16 * j + lane15) * LDST + kl);
    }
    #pragma unroll
    for (int i = 0; i < 2; ++i)
      #pragma unroll
      for (int j = 0; j < 4; ++j) {
        cg[i][j] = __builtin_amdgcn_wmma_f32_16x16x32_bf16(
            false, a[i], false, bg[j], (short)0, cg[i][j], false, false);
        cu[i][j] = __builtin_amdgcn_wmma_f32_16x16x32_bf16(
            false, a[i], false, bu[j], (short)0, cu[i][j], false, false);
      }
  }

  const int rlo = (lane >> 4) * 8;  // C/D layout: lanes>=16 hold M+8
  #pragma unroll
  for (int i = 0; i < 2; ++i)
    #pragma unroll
    for (int j = 0; j < 4; ++j) {
      int rowb = m0 + 32 * wm + 16 * i + rlo;
      int colb = n0 + 64 * wn + 16 * j + lane15;
      #pragma unroll
      for (int v = 0; v < 8; ++v) {
        float g = cg[i][j][v];
        float h = (g * fast_sigmoid(g)) * cu[i][j][v];
        hid[(size_t)(rowb + v) * Hpad + colb] = f32_to_bf16(h);
      }
    }
}

// ---------- GEMM2: out[idx[slot]] += scale[slot] * (hidden[slot] dW^T) ----------

__global__ __launch_bounds__(256) void gemm_down_acc(
    const u16* __restrict__ hid, const u16* __restrict__ dw,
    const int* __restrict__ idxp, const float* __restrict__ gscp,
    const int* __restrict__ cntp, int Hpad, float* __restrict__ out) {
  __shared__ alignas(16) u16 lA[2 * TM * LDST];
  __shared__ alignas(16) u16 lB[2 * TNW * LDST];
  __shared__ float lS[TM];
  __shared__ int lIdx[TM];

  const int tid    = threadIdx.x;
  const int lane   = tid & 31;
  const int wave   = tid >> 5;
  const int wm     = wave & 3;
  const int wn     = wave >> 2;
  const int m0     = blockIdx.y * TM;
  const int n0     = blockIdx.x * TNW;
  const int lane15 = lane & 15;
  const int kl     = (lane < 16) ? 0 : 8;

  const int cnt = cntp ? *cntp : NTOK;
  if (m0 >= cnt) return;

  if (tid < TM) {
    lIdx[tid] = idxp ? idxp[m0 + tid] : (m0 + tid);
    lS[tid]   = gscp ? gscp[m0 + tid] : 1.0f;
  }
  __syncthreads();

  const u16* pA[2]; const u16* pB[2];
  unsigned dA[2], dB[2];
  #pragma unroll
  for (int i = 0; i < 2; ++i) {
    int chunk = tid + 256 * i;
    int r = chunk >> 2, j = chunk & 3;
    pA[i] = hid + (size_t)(m0 + r) * Hpad + 8 * j;   // compacted rows: direct
    pB[i] = dw + (size_t)(n0 + r) * Hpad + 8 * j;
    dA[i] = lds_off(lA + r * LDST + 8 * j);
    dB[i] = lds_off(lB + r * LDST + 8 * j);
  }

  auto fill = [&](int buf, int k0) {
    unsigned bo = buf * BUFB;
    #pragma unroll
    for (int i = 0; i < 2; ++i) {
      async_b128(dA[i] + bo, pA[i] + k0);
      async_b128(dB[i] + bo, pB[i] + k0);
    }
  };

  v8f acc[2][4];
  #pragma unroll
  for (int i = 0; i < 2; ++i)
    #pragma unroll
    for (int j = 0; j < 4; ++j) acc[i][j] = (v8f)0.0f;

  const int KIT = Hpad / 32;
  fill(0, 0);
  for (int it = 0; it < KIT; ++it) {
    wait_async0();
    __syncthreads();
    if (it + 1 < KIT) fill((it + 1) & 1, 32 * (it + 1));

    const u16* bA = lA + (it & 1) * TM * LDST;
    const u16* bB = lB + (it & 1) * TNW * LDST;
    v16bf a[2], b[4];
    #pragma unroll
    for (int i = 0; i < 2; ++i)
      a[i] = ld_frag(bA + (32 * wm + 16 * i + lane15) * LDST + kl);
    #pragma unroll
    for (int j = 0; j < 4; ++j)
      b[j] = ld_frag(bB + (64 * wn + 16 * j + lane15) * LDST + kl);
    #pragma unroll
    for (int i = 0; i < 2; ++i)
      #pragma unroll
      for (int j = 0; j < 4; ++j)
        acc[i][j] = __builtin_amdgcn_wmma_f32_16x16x32_bf16(
            false, a[i], false, b[j], (short)0, acc[i][j], false, false);
  }

  const int rlo = (lane >> 4) * 8;
  #pragma unroll
  for (int i = 0; i < 2; ++i)
    #pragma unroll
    for (int j = 0; j < 4; ++j) {
      int rowb = m0 + 32 * wm + 16 * i + rlo;
      int colb = n0 + 64 * wn + 16 * j + lane15;
      #pragma unroll
      for (int v = 0; v < 8; ++v) {
        int slot = rowb + v;
        if (slot < cnt) {
          int token = lIdx[slot - m0];
          float s = lS[slot - m0];
          out[(size_t)token * CDIM + colb] += s * acc[i][j][v];
        }
      }
    }
}

// ---------- host ----------

extern "C" void kernel_launch(void* const* d_in, const int* in_sizes, int n_in,
                              void* d_out, int out_size, void* d_ws, size_t ws_size,
                              hipStream_t stream) {
  (void)in_sizes; (void)n_in; (void)out_size; (void)ws_size;
  // family order: base, e0, e1, e2
  const int H[4]  = {1152, 1152, 1344, 1536};
  const int HP[4] = {1152, 1152, 1408, 1536};  // padded to multiple of 128

  const float* x    = (const float*)d_in[0];
  const float* rw   = (const float*)d_in[4];
  const float* bias = (const float*)d_in[5];
  const float* gsrc[4] = {(const float*)d_in[1], (const float*)d_in[6],
                          (const float*)d_in[9], (const float*)d_in[12]};
  const float* usrc[4] = {(const float*)d_in[2], (const float*)d_in[7],
                          (const float*)d_in[10], (const float*)d_in[13]};
  const float* dsrc[4] = {(const float*)d_in[3], (const float*)d_in[8],
                          (const float*)d_in[11], (const float*)d_in[14]};

  // workspace carve
  char* base = (char*)d_ws;
  size_t off = 0;
  auto carve = [&](size_t bytes) -> char* {
    char* p = base + off;
    off += (bytes + 255) & ~(size_t)255;
    return p;
  };
  u16* xb = (u16*)carve((size_t)NTOK * CDIM * 2);
  u16 *gwb[4], *uwb[4], *dwb[4];
  for (int f = 0; f < 4; ++f) {
    gwb[f] = (u16*)carve((size_t)HP[f] * CDIM * 2);
    uwb[f] = (u16*)carve((size_t)HP[f] * CDIM * 2);
    dwb[f] = (u16*)carve((size_t)CDIM * HP[f] * 2);
  }
  u16*   hid    = (u16*)carve((size_t)NTOK * 1536 * 2);
  int*   eidx   = (int*)carve((size_t)3 * NTOK * 4);
  float* escale = (float*)carve((size_t)3 * NTOK * 4);
  int*   counts = (int*)carve(256);

  // bf16 conversions (with H padding)
  cvt_f32_bf16<<<1024, 256, 0, stream>>>(x, xb, NTOK * CDIM);
  for (int f = 0; f < 4; ++f) {
    cvt_pad_tail<<<512, 256, 0, stream>>>(gsrc[f], gwb[f], H[f] * CDIM, HP[f] * CDIM);
    cvt_pad_tail<<<512, 256, 0, stream>>>(usrc[f], uwb[f], H[f] * CDIM, HP[f] * CDIM);
    cvt_pad_cols<<<512, 256, 0, stream>>>(dsrc[f], dwb[f], CDIM, H[f], HP[f]);
  }

  // routing lists (zero idx so tail-tile gathers hit a valid token)
  zero_i32<<<1, 64, 0, stream>>>(counts, 3);
  zero_i32<<<512, 256, 0, stream>>>(eidx, 3 * NTOK);
  router_kernel<<<(NTOK * 32) / 256, 256, 0, stream>>>(x, rw, bias, counts, eidx, escale, NTOK);

  // residual init
  copy_f32<<<1024, 256, 0, stream>>>(x, (float*)d_out, NTOK * CDIM);

  // base (all tokens) + experts (compacted, early-exit past count)
  for (int f = 0; f < 4; ++f) {
    const int*   idxp = (f == 0) ? nullptr : (eidx + (size_t)(f - 1) * NTOK);
    const float* gscp = (f == 0) ? nullptr : (escale + (size_t)(f - 1) * NTOK);
    const int*   cntp = (f == 0) ? nullptr : (counts + (f - 1));
    dim3 g1(HP[f] / TNW, NTOK / TM);
    gemm_swiglu<<<g1, 256, 0, stream>>>(xb, gwb[f], uwb[f], hid, idxp, cntp, HP[f]);
    dim3 g2(CDIM / TNW, NTOK / TM);
    gemm_down_acc<<<g2, 256, 0, stream>>>(hid, dwb[f], idxp, gscp, cntp, HP[f],
                                          (float*)d_out);
  }
}
